// Encoder_Attention_59768764891780
// MI455X (gfx1250) — matrix operand order repaired and checked by reading, not compiled, not run on hardware
//
#include <hip/hip_runtime.h>

#define B_    4
#define N_    4096
#define C_    256
#define T_    256
#define KNN_  16
#define EPS_  1e-5f

typedef __bf16 bf16_t;
typedef __attribute__((ext_vector_type(16))) __bf16 v16bf;
typedef __attribute__((ext_vector_type(8)))  float  v8f;

typedef unsigned int u32x4 __attribute__((ext_vector_type(4)));
typedef int          i32x4 __attribute__((ext_vector_type(4)));
typedef int          i32x8 __attribute__((ext_vector_type(8)));

#if defined(__has_builtin)
#if __has_builtin(__builtin_amdgcn_tensor_load_to_lds) && __has_builtin(__builtin_amdgcn_s_wait_tensorcnt)
#define HAVE_TDM 1
#endif
#endif

union BfTile { uint4 q[2]; v16bf v; };

__device__ __forceinline__ bf16_t f2bf(float x) {
    union { float f; unsigned u; } t; t.f = x;
    unsigned r = (t.u + 0x7FFFu + ((t.u >> 16) & 1u)) >> 16;
    union { unsigned short s; bf16_t b; } o; o.s = (unsigned short)r;
    return o.b;
}
__device__ __forceinline__ float bf2f(bf16_t x) {
    union { unsigned short s; bf16_t b; } i; i.b = x;
    union { float f; unsigned u; } t; t.u = ((unsigned)i.s) << 16;
    return t.f;
}

#ifdef HAVE_TDM
// 1-D TDM copy: n8 elements of 8 bytes, contiguous, global -> LDS at byte offset lds_off.
// D# layout per CDNA5 ISA 08_async_tensor §8.3/8.4: group0 {count=1, lds_addr, global_addr, type=2},
// group1 {data_size=3 (8B), tensor_dim0=n8, tensor_dim1=1, tile_dim0=n8, tile_dim1=1, dim0_stride=n8}.
__device__ __forceinline__ void tdm_copy_to_lds(const void* gsrc, unsigned lds_off, unsigned n8) {
    unsigned long long ga = (unsigned long long)gsrc;
    u32x4 g0;
    g0[0] = 1u;                                   // count=1, is_restore=0
    g0[1] = lds_off;                              // lds_addr
    g0[2] = (unsigned)ga;                         // global_addr[31:0]
    g0[3] = (unsigned)((ga >> 32) & 0x01FFFFFFu) | 0x80000000u; // addr[56:32] | type=2
    i32x8 g1;
    g1[0] = (int)(3u << 16);                      // workgroup_mask=0, data_size=3 (8B)
    g1[1] = (int)((n8 & 0xFFFFu) << 16);          // tensor_dim0 low16 in [31:16]
    g1[2] = (int)((n8 >> 16) & 0xFFFFu) | (1 << 16); // tensor_dim0 hi16 | tensor_dim1=1 (low16)
    g1[3] = (int)((n8 & 0xFFFFu) << 16);          // tensor_dim1 hi=0 | tile_dim0 in [31:16]
    g1[4] = 1;                                    // tile_dim1=1, tile_dim2=0
    g1[5] = (int)n8;                              // tensor_dim0_stride low32
    g1[6] = 0;
    g1[7] = 0;
    i32x4 g2 = {0, 0, 0, 0};
    i32x4 g3 = {0, 0, 0, 0};
#if __clang_major__ >= 23
    i32x8 g4 = {0, 0, 0, 0, 0, 0, 0, 0};
    __builtin_amdgcn_tensor_load_to_lds(g0, g1, g2, g3, g4, 0);
#else
    __builtin_amdgcn_tensor_load_to_lds(g0, g1, g2, g3, 0);
#endif
}
#endif

// A-matrix 16x32 bf16 tile from row-major [m][k] (stride in elements).
__device__ __forceinline__ v16bf load_tileA(const bf16_t* base, int stride, int m0, int k0, int lane) {
    int row = m0 + (lane & 15);
    int kb  = k0 + ((lane >> 4) << 3);
    BfTile t;
    t.q[0] = *(const uint4*)(base + (size_t)row * stride + kb);
    t.q[1] = *(const uint4*)(base + (size_t)row * stride + kb + 16);
    return t.v;
}
// B-matrix 32x16 bf16 tile from row-major [n][k].
__device__ __forceinline__ v16bf load_tileB(const bf16_t* base, int stride, int k0, int lane) {
    int row = lane & 15;
    int kb  = k0 + ((lane >> 4) << 3);
    BfTile t;
    t.q[0] = *(const uint4*)(base + (size_t)row * stride + kb);
    t.q[1] = *(const uint4*)(base + (size_t)row * stride + kb + 16);
    return t.v;
}
__device__ __forceinline__ v8f wmma_bf16(v16bf a, v16bf b, v8f c) {
    return __builtin_amdgcn_wmma_f32_16x16x32_bf16(false, a, false, b, (short)0, c, false, false);
}

// ---------------- kernel 0: fp32 -> bf16 weight pack ----------------
__global__ void k_cvt_bf16(const float* __restrict__ src, bf16_t* __restrict__ dst, int n) {
    int i = blockIdx.x * blockDim.x + threadIdx.x;
    if (i < n) dst[i] = f2bf(src[i]);
}

// ---------------- kernel: time gate / bias (B x C) ----------------
__global__ void k_gate(const float* __restrict__ te, const float* __restrict__ gw,
                       const float* __restrict__ gb, const float* __restrict__ hw,
                       float* __restrict__ gate, float* __restrict__ hb) {
    int i = blockIdx.x * blockDim.x + threadIdx.x;   // 0..1023
    int b = i >> 8, c = i & 255;
    const float* t = te + b * T_;
    const float* g = gw + c * T_;
    const float* h = hw + c * T_;
    float sg = 0.f, sh = 0.f;
    for (int k = 0; k < T_; ++k) { sg += g[k] * t[k]; sh += h[k] * t[k]; }
    sg += gb[c];
    gate[i] = 1.0f / (1.0f + __expf(-sg));
    hb[i] = sh;
}

// ---------------- kernel 1: QKV projections (WMMA bf16) ----------------
__global__ void k_qkv(const float* __restrict__ feats, const bf16_t* __restrict__ wbf,
                      const float* __restrict__ qb, const float* __restrict__ kbia,
                      const float* __restrict__ vb,
                      float* __restrict__ qo, float* __restrict__ ko, float* __restrict__ vo) {
    extern __shared__ char smem[];
    bf16_t* wbuf = (bf16_t*)smem;             // 256x256 bf16 (128KB), A layout [m][k]
    bf16_t* fbuf = (bf16_t*)(smem + 131072);  // 128x256 bf16 (64KB),  B layout [n][k]
    int t = threadIdx.x;
    int b  = blockIdx.x >> 5;
    int n0 = (blockIdx.x & 31) << 7;

    for (int i = 0; i < 128; ++i) {
        int e = i * 256 + t;
        int nl = e & 127, c = e >> 7;
        fbuf[nl * 256 + c] = f2bf(feats[((size_t)(b * 256 + c)) * 4096 + n0 + nl]);
    }
    __syncthreads();

    int wave = t >> 5, lane = t & 31;
    int nw = wave << 4;
    int half8 = (lane >> 4) << 3;
    for (int p = 0; p < 3; ++p) {
        // stage 128KB of weights into LDS: TDM descriptor DMA when available
#ifdef HAVE_TDM
        if (t < 32) {
            tdm_copy_to_lds(wbf + p * 65536, 0u, 16384u);
            __builtin_amdgcn_s_wait_tensorcnt(0);
        }
#else
        { const uint4* src = (const uint4*)(wbf + p * 65536);
          uint4* dst = (uint4*)wbuf;
          for (int i = t; i < 8192; i += 256) dst[i] = src[i]; }
#endif
        __syncthreads();
        const float* bias = (p == 0) ? qb : ((p == 1) ? kbia : vb);
        float* outp = (p == 0) ? qo : ((p == 1) ? ko : vo);
        #pragma unroll 1
        for (int mt = 0; mt < 16; ++mt) {
            v8f acc = {0.f,0.f,0.f,0.f,0.f,0.f,0.f,0.f};
            #pragma unroll
            for (int kc = 0; kc < 8; ++kc) {
                v16bf a  = load_tileA(wbuf, 256, mt * 16, kc * 32, lane);
                v16bf bm = load_tileB(fbuf + nw * 256, 256, kc * 32, lane);
                acc = wmma_bf16(a, bm, acc);
            }
            int nn = n0 + nw + (lane & 15);
            #pragma unroll
            for (int r = 0; r < 8; ++r) {
                int ch = mt * 16 + r + half8;
                outp[((size_t)(b * 256 + ch)) * 4096 + nn] = acc[r] + bias[ch];
            }
        }
        __syncthreads();
    }
}

// ---------------- kernel 2: KNN top-16 ----------------
__global__ void k_knn(const float* __restrict__ pts, int* __restrict__ idx) {
    __shared__ float cb[3][1024];
    int t = threadIdx.x;
    int b = blockIdx.x >> 4;
    int n = ((blockIdx.x & 15) << 8) + t;
    const float* px = pts + (size_t)b * 3 * 4096;
    float qx = px[n], qy = px[4096 + n], qz = px[8192 + n];
    float td[16]; int ti[16];
    #pragma unroll
    for (int i = 0; i < 16; ++i) { td[i] = 3.4e38f; ti[i] = 0; }
    for (int c0 = 0; c0 < 4096; c0 += 1024) {
        __syncthreads();
        for (int i = 0; i < 12; ++i) {
            int e = i * 256 + t;
            int g = e >> 10, j = e & 1023;
            cb[g][j] = px[g * 4096 + c0 + j];
        }
        __syncthreads();
        for (int j = 0; j < 1024; ++j) {
            float dx = qx - cb[0][j], dy = qy - cb[1][j], dz = qz - cb[2][j];
            float d = dx * dx + dy * dy + dz * dz;
            if (d < td[15]) {
                td[15] = d; ti[15] = c0 + j;
                #pragma unroll
                for (int s = 15; s > 0; --s) {
                    if (td[s] < td[s - 1]) {
                        float a = td[s]; td[s] = td[s - 1]; td[s - 1] = a;
                        int   c = ti[s]; ti[s] = ti[s - 1]; ti[s - 1] = c;
                    } else break;
                }
            }
        }
    }
    int* op = idx + ((size_t)(b * 4096 + n)) * 16;
    #pragma unroll
    for (int i = 0; i < 16; ++i) op[i] = ti[i];
}

// ---------------- kernel 3: fused gather + geo/rel MLP + softmax + aggregate ----------------
__global__ void k_attn(const float* __restrict__ pts, const float* __restrict__ feats,
                       const float* __restrict__ qws, const float* __restrict__ kws,
                       const float* __restrict__ vws, const int* __restrict__ idx,
                       const bf16_t* __restrict__ g2wb, const bf16_t* __restrict__ r1wb,
                       const bf16_t* __restrict__ r2wb,
                       const float* __restrict__ g1w, const float* __restrict__ g1b,
                       const float* __restrict__ gbn_g, const float* __restrict__ gbn_b,
                       const float* __restrict__ g2b, const float* __restrict__ r1b,
                       const float* __restrict__ rbn_g, const float* __restrict__ rbn_b,
                       const float* __restrict__ r2b, float* __restrict__ agg) {
    extern __shared__ char smem[];
    bf16_t* g2s = (bf16_t*)smem;                 // 256x128
    bf16_t* r1s = (bf16_t*)(smem + 65536);       // 128x256
    bf16_t* r2s = (bf16_t*)(smem + 131072);      // 256x128
    float* g1ws = (float*)(smem + 196608);       // 384
    float* g1sc = g1ws + 384;                    // 128
    float* g1of = g1sc + 128;                    // 128
    float* g2bs = g1of + 128;                    // 256
    float* r1sc = g2bs + 256;                    // 128
    float* r1of = r1sc + 128;                    // 128
    float* r2bs = r1of + 128;                    // 256
    char*  pw   = smem + 196608 + (384 + 128 + 128 + 256 + 128 + 128 + 256) * 4;

    int t = threadIdx.x;
    // stage 192KB of bf16 weights (g2|r1|r2 are contiguous in ws, dest LDS contiguous from offset 0)
#ifdef HAVE_TDM
    if (t < 32) {
        tdm_copy_to_lds(g2wb, 0u, 24576u);
        __builtin_amdgcn_s_wait_tensorcnt(0);
    }
#else
    { const uint4* s0 = (const uint4*)g2wb; uint4* d0 = (uint4*)g2s;
      for (int i = t; i < 4096; i += 128) d0[i] = s0[i];
      const uint4* s1 = (const uint4*)r1wb; uint4* d1 = (uint4*)r1s;
      for (int i = t; i < 4096; i += 128) d1[i] = s1[i];
      const uint4* s2 = (const uint4*)r2wb; uint4* d2 = (uint4*)r2s;
      for (int i = t; i < 4096; i += 128) d2[i] = s2[i]; }
#endif
    float rs = rsqrtf(1.0f + EPS_);
    for (int i = t; i < 384; i += 128) g1ws[i] = g1w[i];
    for (int i = t; i < 128; i += 128) {
        float s = gbn_g[i] * rs; g1sc[i] = s; g1of[i] = g1b[i] * s + gbn_b[i];
        float s2 = rbn_g[i] * rs; r1sc[i] = s2; r1of[i] = r1b[i] * s2 + rbn_b[i];
    }
    for (int i = t; i < 256; i += 128) { g2bs[i] = g2b[i]; r2bs[i] = r2b[i]; }
    __syncthreads();

    int wave = t >> 5, lane = t & 31;
    bf16_t* geobuf = (bf16_t*)(pw + wave * 20480);           // [ch=256][nbr=16]
    bf16_t* relbuf = (bf16_t*)(pw + wave * 20480 + 8192);    // [nbr=16][ch=256]
    bf16_t* buf1   = (bf16_t*)(pw + wave * 20480 + 16384);   // [nbr=16][ch=128]
    int nbr = lane & 15;
    int half8 = (lane >> 4) << 3;
    int nslots = gridDim.x * 4;

    #pragma unroll 1
    for (int p = blockIdx.x * 4 + wave; p < B_ * N_; p += nslots) {
        int b = p >> 12, n = p & 4095;
        const float* px = pts + (size_t)b * 3 * 4096;
        int myidx = idx[(size_t)p * 16 + nbr];
        float dxl = px[n]        - px[myidx];
        float dyl = px[4096 + n] - px[4096 + myidx];
        float dzl = px[8192 + n] - px[8192 + myidx];

        // geo1 (3 -> 128), VALU; result in B layout buf1[nbr][c]
        #pragma unroll 4
        for (int i = 0; i < 64; ++i) {
            int c = (lane >> 4) + (i << 1);
            float v0 = g1ws[c * 3 + 0] * dxl + g1ws[c * 3 + 1] * dyl + g1ws[c * 3 + 2] * dzl;
            v0 = fmaxf(v0 * g1sc[c] + g1of[c], 0.0f);
            buf1[nbr * 128 + c] = f2bf(v0);
        }

        const float* qp = qws + (size_t)b * 256 * 4096;
        const float* kp = kws + (size_t)b * 256 * 4096;
        // geo2 (128 -> 256) + build rel_in = q - knn_k + geo
        #pragma unroll 1
        for (int mt = 0; mt < 16; ++mt) {
            v8f acc = {0.f,0.f,0.f,0.f,0.f,0.f,0.f,0.f};
            #pragma unroll
            for (int kc = 0; kc < 4; ++kc) {
                v16bf a  = load_tileA(g2s, 128, mt * 16, kc * 32, lane);
                v16bf bm = load_tileB(buf1, 128, kc * 32, lane);
                acc = wmma_bf16(a, bm, acc);
            }
            #pragma unroll
            for (int r = 0; r < 8; ++r) {
                int ch = mt * 16 + r + half8;
                float gv = acc[r] + g2bs[ch];
                geobuf[ch * 16 + nbr] = f2bf(gv);
                float qv = qp[(size_t)ch * 4096 + n];
                float kv = kp[(size_t)ch * 4096 + myidx];
                relbuf[nbr * 256 + ch] = f2bf(qv - kv + gv);
            }
        }
        // rel1 (256 -> 128), relu+bn, into buf1
        #pragma unroll 1
        for (int mt = 0; mt < 8; ++mt) {
            v8f acc = {0.f,0.f,0.f,0.f,0.f,0.f,0.f,0.f};
            #pragma unroll
            for (int kc = 0; kc < 8; ++kc) {
                v16bf a  = load_tileA(r1s, 256, mt * 16, kc * 32, lane);
                v16bf bm = load_tileB(relbuf, 256, kc * 32, lane);
                acc = wmma_bf16(a, bm, acc);
            }
            #pragma unroll
            for (int r = 0; r < 8; ++r) {
                int ch = mt * 16 + r + half8;
                float v1 = fmaxf(acc[r] * r1sc[ch] + r1of[ch], 0.0f);
                buf1[nbr * 128 + ch] = f2bf(v1);
            }
        }
        // rel2 (128 -> 256) + softmax over neighbors + aggregate
        const float* vp = vws + (size_t)b * 256 * 4096;
        const float* fp = feats + (size_t)b * 256 * 4096;
        float* ap = agg + (size_t)b * 256 * 4096;
        #pragma unroll 1
        for (int mt = 0; mt < 16; ++mt) {
            v8f acc = {0.f,0.f,0.f,0.f,0.f,0.f,0.f,0.f};
            #pragma unroll
            for (int kc = 0; kc < 4; ++kc) {
                v16bf a  = load_tileA(r2s, 128, mt * 16, kc * 32, lane);
                v16bf bm = load_tileB(buf1, 128, kc * 32, lane);
                acc = wmma_bf16(a, bm, acc);
            }
            #pragma unroll
            for (int r = 0; r < 8; ++r) {
                int ch = mt * 16 + r + half8;
                float x = acc[r] + r2bs[ch];
                float m = x;
                m = fmaxf(m, __shfl_xor(m, 1));
                m = fmaxf(m, __shfl_xor(m, 2));
                m = fmaxf(m, __shfl_xor(m, 4));
                m = fmaxf(m, __shfl_xor(m, 8));
                float e = __expf(x - m);
                float s = e;
                s += __shfl_xor(s, 1);
                s += __shfl_xor(s, 2);
                s += __shfl_xor(s, 4);
                s += __shfl_xor(s, 8);
                float attn = e / s;
                float vv = vp[(size_t)ch * 4096 + myidx] + bf2f(geobuf[ch * 16 + nbr]);
                float contrib = attn * vv;
                contrib += __shfl_xor(contrib, 1);
                contrib += __shfl_xor(contrib, 2);
                contrib += __shfl_xor(contrib, 4);
                contrib += __shfl_xor(contrib, 8);
                if (nbr == 0)
                    ap[(size_t)ch * 4096 + n] = contrib + fp[(size_t)ch * 4096 + n];
            }
        }
    }
}

// ---------------- kernel 4: output head (o1 -> o2 -> residual -> tw, gate) ----------------
__global__ void k_head(const float* __restrict__ aggp,
                       const bf16_t* __restrict__ o1wb, const bf16_t* __restrict__ o2wb,
                       const bf16_t* __restrict__ twb,
                       const float* __restrict__ o1b, const float* __restrict__ obn1_g,
                       const float* __restrict__ obn1_b,
                       const float* __restrict__ o2b, const float* __restrict__ obn2_g,
                       const float* __restrict__ obn2_b,
                       const float* __restrict__ tb,
                       const float* __restrict__ gate, const float* __restrict__ hbias,
                       float* __restrict__ out) {
    extern __shared__ char smem[];
    float* o1sc = (float*)smem;       // 512
    float* o1of = o1sc + 512;         // 512
    float* o2sc = o1of + 512;         // 256
    float* o2of = o2sc + 256;         // 256
    float* tbs  = o2of + 256;         // 256
    char*  pw   = (char*)(tbs + 256);

    int t = threadIdx.x;
    float rs = rsqrtf(1.0f + EPS_);
    for (int i = t; i < 512; i += 128) { float s = obn1_g[i] * rs; o1sc[i] = s; o1of[i] = o1b[i] * s + obn1_b[i]; }
    for (int i = t; i < 256; i += 128) { float s = obn2_g[i] * rs; o2sc[i] = s; o2of[i] = o2b[i] * s + obn2_b[i]; tbs[i] = tb[i]; }
    __syncthreads();

    int wave = t >> 5, lane = t & 31;
    bf16_t* aggbuf = (bf16_t*)(pw + wave * 32768);            // [n=16][256]
    bf16_t* hbuf   = (bf16_t*)(pw + wave * 32768 + 8192);     // [n=16][512]
    bf16_t* ofbuf  = (bf16_t*)(pw + wave * 32768 + 24576);    // [n=16][256]
    int nbr = lane & 15, half8 = (lane >> 4) << 3;
    int tiles = (B_ * N_) / 16;

    #pragma unroll 1
    for (int tt = blockIdx.x * 4 + wave; tt < tiles; tt += gridDim.x * 4) {
        int b = tt >> 8;
        int n0 = (tt & 255) << 4;
        const float* ag = aggp + (size_t)b * 256 * 4096;
        for (int i = 0; i < 128; ++i) {
            int e = i * 32 + lane;
            int nl = e >> 8, c = e & 255;
            aggbuf[nl * 256 + c] = f2bf(ag[(size_t)c * 4096 + n0 + nl]);
        }
        // o1: 256 -> 512, relu+bn
        #pragma unroll 1
        for (int mt = 0; mt < 32; ++mt) {
            if (mt < 31)
                __builtin_prefetch(o1wb + (size_t)(mt * 16 + 16 + (lane & 15)) * 256, 0, 1);
            v8f acc = {0.f,0.f,0.f,0.f,0.f,0.f,0.f,0.f};
            #pragma unroll
            for (int kc = 0; kc < 8; ++kc) {
                v16bf a  = load_tileA(o1wb, 256, mt * 16, kc * 32, lane);
                v16bf bm = load_tileB(aggbuf, 256, kc * 32, lane);
                acc = wmma_bf16(a, bm, acc);
            }
            #pragma unroll
            for (int r = 0; r < 8; ++r) {
                int ch = mt * 16 + r + half8;
                float h = fmaxf(acc[r] * o1sc[ch] + o1of[ch], 0.0f);
                hbuf[nbr * 512 + ch] = f2bf(h);
            }
        }
        // o2: 512 -> 256, bn, + residual agg
        #pragma unroll 1
        for (int mt = 0; mt < 16; ++mt) {
            if (mt < 15)
                __builtin_prefetch(o2wb + (size_t)(mt * 16 + 16 + (lane & 15)) * 512, 0, 1);
            v8f acc = {0.f,0.f,0.f,0.f,0.f,0.f,0.f,0.f};
            #pragma unroll
            for (int kc = 0; kc < 16; ++kc) {
                v16bf a  = load_tileA(o2wb, 512, mt * 16, kc * 32, lane);
                v16bf bm = load_tileB(hbuf, 512, kc * 32, lane);
                acc = wmma_bf16(a, bm, acc);
            }
            #pragma unroll
            for (int r = 0; r < 8; ++r) {
                int ch = mt * 16 + r + half8;
                float h2 = acc[r] * o2sc[ch] + o2of[ch];
                float of = h2 + ag[(size_t)ch * 4096 + n0 + nbr];
                ofbuf[nbr * 256 + ch] = f2bf(of);
            }
        }
        // tw: 256 -> 256, * gate + hbias
        const float* gp = gate + b * 256;
        const float* hp = hbias + b * 256;
        #pragma unroll 1
        for (int mt = 0; mt < 16; ++mt) {
            if (mt < 15)
                __builtin_prefetch(twb + (size_t)(mt * 16 + 16 + (lane & 15)) * 256, 0, 1);
            v8f acc = {0.f,0.f,0.f,0.f,0.f,0.f,0.f,0.f};
            #pragma unroll
            for (int kc = 0; kc < 8; ++kc) {
                v16bf a  = load_tileA(twb, 256, mt * 16, kc * 32, lane);
                v16bf bm = load_tileB(ofbuf, 256, kc * 32, lane);
                acc = wmma_bf16(a, bm, acc);
            }
            #pragma unroll
            for (int r = 0; r < 8; ++r) {
                int ch = mt * 16 + r + half8;
                float v0 = acc[r] + tbs[ch];
                out[((size_t)(b * 256 + ch)) * 4096 + n0 + nbr] = v0 * gp[ch] + hp[ch];
            }
        }
    }
}

extern "C" void kernel_launch(void* const* d_in, const int* in_sizes, int n_in,
                              void* d_out, int out_size, void* d_ws, size_t ws_size,
                              hipStream_t stream) {
    const float* pts    = (const float*)d_in[0];
    const float* feats  = (const float*)d_in[1];
    const float* time_e = (const float*)d_in[2];
    const float* qw = (const float*)d_in[3];   const float* qb = (const float*)d_in[4];
    const float* kw = (const float*)d_in[5];   const float* kb = (const float*)d_in[6];
    const float* vw = (const float*)d_in[7];   const float* vb = (const float*)d_in[8];
    const float* g1w = (const float*)d_in[9];  const float* g1b = (const float*)d_in[10];
    const float* gbn_g = (const float*)d_in[11]; const float* gbn_b = (const float*)d_in[12];
    const float* g2w = (const float*)d_in[13]; const float* g2b = (const float*)d_in[14];
    const float* r1w = (const float*)d_in[15]; const float* r1b = (const float*)d_in[16];
    const float* rbn_g = (const float*)d_in[17]; const float* rbn_b = (const float*)d_in[18];
    const float* r2w = (const float*)d_in[19]; const float* r2b = (const float*)d_in[20];
    const float* o1w = (const float*)d_in[21]; const float* o1b = (const float*)d_in[22];
    const float* obn1_g = (const float*)d_in[23]; const float* obn1_b = (const float*)d_in[24];
    const float* o2w = (const float*)d_in[25]; const float* o2b = (const float*)d_in[26];
    const float* obn2_g = (const float*)d_in[27]; const float* obn2_b = (const float*)d_in[28];
    const float* tw = (const float*)d_in[29];  const float* tb = (const float*)d_in[30];
    const float* gate_w = (const float*)d_in[31]; const float* gate_b = (const float*)d_in[32];
    const float* hb_w = (const float*)d_in[33];

    char* ws = (char*)d_ws;
    const size_t MB16 = (size_t)1 << 24;
    float* q_ws   = (float*)(ws);
    float* k_ws   = (float*)(ws + MB16);
    float* v_ws   = (float*)(ws + 2 * MB16);
    float* agg_ws = (float*)(ws + 3 * MB16);
    int*   idx_ws = (int*)  (ws + 4 * MB16);
    float* gate_ws = (float*)(ws + 4 * MB16 + (1 << 20));
    float* hb_ws   = gate_ws + 1024;
    bf16_t* wbf    = (bf16_t*)(ws + 4 * MB16 + (1 << 20) + 16384);
    // bf16 weight offsets (elements)
    const int QW = 0, KW = 65536, VW = 131072, G2 = 196608, R1 = 229376, R2 = 262144,
              O1 = 294912, O2 = 425984, TW = 557056;

    k_cvt_bf16<<<256, 256, 0, stream>>>(qw,  wbf + QW, 65536);
    k_cvt_bf16<<<256, 256, 0, stream>>>(kw,  wbf + KW, 65536);
    k_cvt_bf16<<<256, 256, 0, stream>>>(vw,  wbf + VW, 65536);
    k_cvt_bf16<<<128, 256, 0, stream>>>(g2w, wbf + G2, 32768);
    k_cvt_bf16<<<128, 256, 0, stream>>>(r1w, wbf + R1, 32768);
    k_cvt_bf16<<<128, 256, 0, stream>>>(r2w, wbf + R2, 32768);
    k_cvt_bf16<<<512, 256, 0, stream>>>(o1w, wbf + O1, 131072);
    k_cvt_bf16<<<512, 256, 0, stream>>>(o2w, wbf + O2, 131072);
    k_cvt_bf16<<<256, 256, 0, stream>>>(tw,  wbf + TW, 65536);

    k_gate<<<4, 256, 0, stream>>>(time_e, gate_w, gate_b, hb_w, gate_ws, hb_ws);
    k_qkv<<<128, 256, 196608, stream>>>(feats, wbf, qb, kb, vb, q_ws, k_ws, v_ws);
    k_knn<<<64, 256, 0, stream>>>(pts, idx_ws);
    k_attn<<<512, 128, 284160, stream>>>(pts, feats, q_ws, k_ws, v_ws, idx_ws,
        wbf + G2, wbf + R1, wbf + R2,
        g1w, g1b, gbn_g, gbn_b, g2b, r1b, rbn_g, rbn_b, r2b, agg_ws);
    k_head<<<256, 128, 138240, stream>>>(agg_ws, wbf + O1, wbf + O2, wbf + TW,
        o1b, obn1_g, obn1_b, o2b, obn2_g, obn2_b, tb, gate_ws, hb_ws, (float*)d_out);
}
